// MultiHeadAttention_57999238365449
// MI455X (gfx1250) — matrix operand order repaired and checked by reading, not compiled, not run on hardware
//
#include <hip/hip_runtime.h>

#define Bb   2
#define Ss   2048
#define Dd   1024
#define Hh   16
#define DHh  64
#define BS   (Bb * Ss)      // 4096

typedef __attribute__((ext_vector_type(16))) _Float16 v16h;
typedef __attribute__((ext_vector_type(8)))  _Float16 v8h;
typedef __attribute__((ext_vector_type(4)))  _Float16 v4h;
typedef __attribute__((ext_vector_type(8)))  float    v8f;

__device__ __forceinline__ v8f wmma16(v16h a, v16h b, v8f c) {
    // emits v_wmma_f32_16x16x32_f16
    return __builtin_amdgcn_wmma_f32_16x16x32_f16(
        false, a, false, b, (short)0, c, false, false);
}

__device__ __forceinline__ v8f zero8() {
    v8f c;
#pragma unroll
    for (int i = 0; i < 8; ++i) c[i] = 0.0f;
    return c;
}

// A fragment (16x32 f16): two contiguous 8-element chunks per lane, fp32 source.
__device__ __forceinline__ v16h load_a_f32(const float* __restrict__ row, int off0, int off1) {
    const float4* p0 = (const float4*)(row + off0);
    const float4* p1 = (const float4*)(row + off1);
    float4 f0 = p0[0], f1 = p0[1], f2 = p1[0], f3 = p1[1];
    v16h a;
    a[0]=(_Float16)f0.x; a[1]=(_Float16)f0.y; a[2]=(_Float16)f0.z; a[3]=(_Float16)f0.w;
    a[4]=(_Float16)f1.x; a[5]=(_Float16)f1.y; a[6]=(_Float16)f1.z; a[7]=(_Float16)f1.w;
    a[8]=(_Float16)f2.x; a[9]=(_Float16)f2.y; a[10]=(_Float16)f2.z; a[11]=(_Float16)f2.w;
    a[12]=(_Float16)f3.x; a[13]=(_Float16)f3.y; a[14]=(_Float16)f3.z; a[15]=(_Float16)f3.w;
    return a;
}

// A fragment from f16 source (two contiguous v8h chunks).
__device__ __forceinline__ v16h load_a_f16(const _Float16* __restrict__ row, int off0, int off1) {
    v8h lo = *(const v8h*)(row + off0);
    v8h hi = *(const v8h*)(row + off1);
    v16h a;
#pragma unroll
    for (int i = 0; i < 8; ++i) { a[i] = lo[i]; a[8 + i] = hi[i]; }
    return a;
}

// B fragment (32x16): two 8-element f16 chunks per lane, at p and p+16 (global or LDS; inlined so
// clang infers the address space -> ds_load_b128 for shared pointers).
__device__ __forceinline__ v16h load_b_f16(const _Float16* p) {
    v8h lo = *(const v8h*)p;
    v8h hi = *(const v8h*)(p + 16);
    v16h b;
#pragma unroll
    for (int i = 0; i < 8; ++i) { b[i] = lo[i]; b[8 + i] = hi[i]; }
    return b;
}

// Cooperative stage of a 64(n) x 32(k) fp32 W tile into LDS as f16.
// 256 threads: thread t -> row t>>2, 8 cols at (t&3)*8. One cvt per element per WG.
__device__ __forceinline__ void stage_w_tile(const float* __restrict__ W, int nbase, int kc,
                                             _Float16* dst, int tid) {
    const int r  = tid >> 2;
    const int c8 = (tid & 3) * 8;
    const float4* src = (const float4*)(W + (size_t)(nbase + r) * Dd + kc + c8);
    float4 f0 = src[0], f1 = src[1];
    v8h h;
    h[0]=(_Float16)f0.x; h[1]=(_Float16)f0.y; h[2]=(_Float16)f0.z; h[3]=(_Float16)f0.w;
    h[4]=(_Float16)f1.x; h[5]=(_Float16)f1.y; h[6]=(_Float16)f1.z; h[7]=(_Float16)f1.w;
    *(v8h*)(dst + r * 32 + c8) = h;
}

// Cooperative stage of a 64(row) x 32(col) f16 tile (row stride = rstride halves).
__device__ __forceinline__ void stage_h_tile64x32(const _Float16* __restrict__ src0,
                                                  size_t rstride, int coff,
                                                  _Float16* dst, int tid) {
    const int r  = tid >> 2;
    const int c8 = (tid & 3) * 8;
    *(v8h*)(dst + r * 32 + c8) = *(const v8h*)(src0 + (size_t)r * rstride + coff + c8);
}

// ---------------------------------------------------------------------------
// K1: Y = X @ W^T + bias, fp32 -> f16, output in [B,H,S,DH] (vtrans=0) or
//     [B,H,DH,S] (vtrans=1, for V).  Grid: (D/64, BS/128), block 256.
// ---------------------------------------------------------------------------
__global__ __launch_bounds__(256) void proj_kernel(
    const float* __restrict__ X, const float* __restrict__ W,
    const float* __restrict__ bias, _Float16* __restrict__ Y, int vtrans)
{
    __shared__ _Float16 Bs[2][64 * 32];
    const int tid   = threadIdx.x;
    const int lane  = tid & 31;
    const int wv    = tid >> 5;
    const int l15   = lane & 15;
    const int lhalf = lane >> 4;
    const int mbase = blockIdx.y * 128 + wv * 16;
    const int nbase = blockIdx.x * 64;

    const float* Xrow = X + (size_t)(mbase + l15) * Dd;

    v8f acc[4];
#pragma unroll
    for (int nt = 0; nt < 4; ++nt) acc[nt] = zero8();

    stage_w_tile(W, nbase, 0, Bs[0], tid);
    __syncthreads();

    const int NIT = Dd / 32;
    for (int i = 0; i < NIT; ++i) {
        const int kc = i * 32;
        if (i + 1 < NIT) stage_w_tile(W, nbase, kc + 32, Bs[(i + 1) & 1], tid);
        v16h a = load_a_f32(Xrow, kc + 8 * lhalf, kc + 16 + 8 * lhalf);
        const _Float16* bs = Bs[i & 1];
        v16h bfr[4];
#pragma unroll
        for (int nt = 0; nt < 4; ++nt)
            bfr[nt] = load_b_f16(bs + (nt * 16 + l15) * 32 + 8 * lhalf);
#pragma unroll
        for (int nt = 0; nt < 4; ++nt)
            acc[nt] = wmma16(a, bfr[nt], acc[nt]);
        __syncthreads();
    }
#pragma unroll
    for (int nt = 0; nt < 4; ++nt) {
        const int n  = nbase + nt * 16 + l15;
        const float bvv = bias[n];
        const int h = n >> 6, dh = n & 63;
#pragma unroll
        for (int r = 0; r < 8; ++r) {
            const int m = mbase + r + 8 * lhalf;
            const int bidx = m >> 11;          // S = 2048
            const int s = m & (Ss - 1);
            const float val = acc[nt][r] + bvv;
            const size_t idx = vtrans
                ? ((size_t)(bidx * Hh + h) * DHh + dh) * Ss + s
                : ((size_t)(bidx * Hh + h) * Ss + s) * DHh + dh;
            Y[idx] = (_Float16)val;
        }
    }
}

// ---------------------------------------------------------------------------
// K2: raw scores = scale * Q K^T streamed to attn buffer; online softmax stats.
//     K tile (16 k-rows x 64 dh) staged in LDS per step.  Grid: (S/128, B*H).
// ---------------------------------------------------------------------------
__global__ __launch_bounds__(256) void scores_kernel(
    const _Float16* __restrict__ Q, const _Float16* __restrict__ Kg,
    float* __restrict__ attn, float* __restrict__ stMax, float* __restrict__ stInv)
{
    __shared__ _Float16 Ks[2][16 * 64];
    const int tid   = threadIdx.x;
    const int lane  = tid & 31;
    const int wv    = tid >> 5;
    const int l15   = lane & 15;
    const int lhalf = lane >> 4;
    const int bh    = blockIdx.y;
    const int qb    = blockIdx.x * 128 + wv * 16;

    const _Float16* Qrow = Q + ((size_t)bh * Ss + qb + l15) * DHh;
    const v16h aq0 = load_a_f16(Qrow,       8 * lhalf, 16 + 8 * lhalf);  // dh 0..31
    const v16h aq1 = load_a_f16(Qrow, 32 +  8 * lhalf, 48 + 8 * lhalf);  // dh 32..63

    // staging thread map: row = tid>>4 (0..15), 4 halves at (tid&15)*4
    const int srow = tid >> 4;
    const int sc4  = (tid & 15) * 4;
    const _Float16* Kbase = Kg + ((size_t)bh * Ss + srow) * DHh + sc4;

    float mrun[8], srun[8];
#pragma unroll
    for (int r = 0; r < 8; ++r) { mrun[r] = -3.0e38f; srun[r] = 0.0f; }

    const float scale = 0.125f;  // 1/sqrt(64)
    float* arow = attn + ((size_t)bh * Ss + qb) * Ss;

    *(v4h*)(&Ks[0][srow * 64 + sc4]) = *(const v4h*)Kbase;
    __syncthreads();

    const int NIT = Ss / 16;
    for (int kt = 0; kt < NIT; ++kt) {
        if (kt + 1 < NIT)
            *(v4h*)(&Ks[(kt + 1) & 1][srow * 64 + sc4]) =
                *(const v4h*)(Kbase + (size_t)(kt + 1) * 16 * DHh);
        const _Float16* ks = Ks[kt & 1];
        v16h b0 = load_b_f16(ks + l15 * 64 + 8 * lhalf);        // dh 0..31
        v16h b1 = load_b_f16(ks + l15 * 64 + 32 + 8 * lhalf);   // dh 32..63
        v8f c = zero8();
        c = wmma16(aq0, b0, c);
        c = wmma16(aq1, b1, c);
        const int col = kt * 16 + l15;
#pragma unroll
        for (int r = 0; r < 8; ++r) {
            const float v = c[r] * scale;
            arow[(size_t)(r + 8 * lhalf) * Ss + col] = v;   // raw score out
            const float nm = fmaxf(mrun[r], v);
            srun[r] = srun[r] * __expf(mrun[r] - nm) + __expf(v - nm);
            mrun[r] = nm;
        }
        __syncthreads();
    }
    // merge 16 per-lane partials per row (rows 0-7 in lanes 0-15, 8-15 in 16-31)
#pragma unroll
    for (int r = 0; r < 8; ++r) {
        float m = mrun[r], s = srun[r];
#pragma unroll
        for (int off = 1; off < 16; off <<= 1) {
            const float om = __shfl_xor(m, off, 32);
            const float os = __shfl_xor(s, off, 32);
            const float nm = fmaxf(m, om);
            s = s * __expf(m - nm) + os * __expf(om - nm);
            m = nm;
        }
        mrun[r] = m; srun[r] = s;
    }
    if (l15 == 0) {
#pragma unroll
        for (int r = 0; r < 8; ++r) {
            const int row = qb + r + 8 * lhalf;
            stMax[(size_t)bh * Ss + row] = mrun[r];
            stInv[(size_t)bh * Ss + row] = 1.0f / srun[r];
        }
    }
}

// ---------------------------------------------------------------------------
// K3: normalize attn in place (final attn output) and O_head = P @ V via WMMA.
//     V^T tile (64 dh x 32 k) staged in LDS per step.  Grid: (S/128, B*H).
// ---------------------------------------------------------------------------
__global__ __launch_bounds__(256) void attnv_kernel(
    float* __restrict__ attn, const _Float16* __restrict__ Vt,
    const float* __restrict__ stMax, const float* __restrict__ stInv,
    _Float16* __restrict__ O)
{
    __shared__ _Float16 Vs[2][64 * 32];
    const int tid   = threadIdx.x;
    const int lane  = tid & 31;
    const int wv    = tid >> 5;
    const int l15   = lane & 15;
    const int lhalf = lane >> 4;
    const int bh    = blockIdx.y;
    const int qb    = blockIdx.x * 128 + wv * 16;

    const int q = qb + l15;                 // A-fragment row for this lane
    const float mx  = stMax[(size_t)bh * Ss + q];
    const float inv = stInv[(size_t)bh * Ss + q];
    float* arow = attn + ((size_t)bh * Ss + q) * Ss;
    const _Float16* Vbase = Vt + (size_t)bh * DHh * Ss;   // [dh][s], row stride Ss

    v8f o[4];
#pragma unroll
    for (int nt = 0; nt < 4; ++nt) o[nt] = zero8();

    stage_h_tile64x32(Vbase, Ss, 0, Vs[0], tid);
    __syncthreads();

    const int NIT = Ss / 32;
    for (int i = 0; i < NIT; ++i) {
        const int kc = i * 32;
        if (i + 1 < NIT) stage_h_tile64x32(Vbase, Ss, kc + 32, Vs[(i + 1) & 1], tid);

        const int o0 = kc + 8 * lhalf;
        const int o1 = kc + 16 + 8 * lhalf;
        float4 f0 = *(const float4*)(arow + o0);
        float4 f1 = *(const float4*)(arow + o0 + 4);
        float4 f2 = *(const float4*)(arow + o1);
        float4 f3 = *(const float4*)(arow + o1 + 4);
        float p[16];
        p[0]=__expf(f0.x-mx)*inv; p[1]=__expf(f0.y-mx)*inv; p[2]=__expf(f0.z-mx)*inv; p[3]=__expf(f0.w-mx)*inv;
        p[4]=__expf(f1.x-mx)*inv; p[5]=__expf(f1.y-mx)*inv; p[6]=__expf(f1.z-mx)*inv; p[7]=__expf(f1.w-mx)*inv;
        p[8]=__expf(f2.x-mx)*inv; p[9]=__expf(f2.y-mx)*inv; p[10]=__expf(f2.z-mx)*inv; p[11]=__expf(f2.w-mx)*inv;
        p[12]=__expf(f3.x-mx)*inv; p[13]=__expf(f3.y-mx)*inv; p[14]=__expf(f3.z-mx)*inv; p[15]=__expf(f3.w-mx)*inv;
        // normalized probabilities are the final attn output
        *(float4*)(arow + o0)     = make_float4(p[0], p[1], p[2], p[3]);
        *(float4*)(arow + o0 + 4) = make_float4(p[4], p[5], p[6], p[7]);
        *(float4*)(arow + o1)     = make_float4(p[8], p[9], p[10], p[11]);
        *(float4*)(arow + o1 + 4) = make_float4(p[12], p[13], p[14], p[15]);

        v16h ap;
#pragma unroll
        for (int j = 0; j < 16; ++j) ap[j] = (_Float16)p[j];

        const _Float16* vs = Vs[i & 1];
        v16h bfr[4];
#pragma unroll
        for (int nt = 0; nt < 4; ++nt)
            bfr[nt] = load_b_f16(vs + (nt * 16 + l15) * 32 + 8 * lhalf);
#pragma unroll
        for (int nt = 0; nt < 4; ++nt)
            o[nt] = wmma16(ap, bfr[nt], o[nt]);
        __syncthreads();
    }
    const int bidx = bh >> 4, h = bh & 15;
#pragma unroll
    for (int nt = 0; nt < 4; ++nt) {
#pragma unroll
        for (int r = 0; r < 8; ++r) {
            const int qrow = qb + r + 8 * lhalf;
            O[(size_t)(bidx * Ss + qrow) * Dd + h * 64 + nt * 16 + l15] = (_Float16)o[nt][r];
        }
    }
}

// ---------------------------------------------------------------------------
// K4: out = O @ Wo^T + bo (f16 A, W staged fp32->f16 in LDS, fp32 out).
//     Grid: (D/64, BS/128), block 256.
// ---------------------------------------------------------------------------
__global__ __launch_bounds__(256) void outproj_kernel(
    const _Float16* __restrict__ Oi, const float* __restrict__ W,
    const float* __restrict__ bias, float* __restrict__ out)
{
    __shared__ _Float16 Bs[2][64 * 32];
    const int tid   = threadIdx.x;
    const int lane  = tid & 31;
    const int wv    = tid >> 5;
    const int l15   = lane & 15;
    const int lhalf = lane >> 4;
    const int mbase = blockIdx.y * 128 + wv * 16;
    const int nbase = blockIdx.x * 64;

    const _Float16* Orow = Oi + (size_t)(mbase + l15) * Dd;

    v8f acc[4];
#pragma unroll
    for (int nt = 0; nt < 4; ++nt) acc[nt] = zero8();

    stage_w_tile(W, nbase, 0, Bs[0], tid);
    __syncthreads();

    const int NIT = Dd / 32;
    for (int i = 0; i < NIT; ++i) {
        const int kc = i * 32;
        if (i + 1 < NIT) stage_w_tile(W, nbase, kc + 32, Bs[(i + 1) & 1], tid);
        v16h a = load_a_f16(Orow, kc + 8 * lhalf, kc + 16 + 8 * lhalf);
        const _Float16* bs = Bs[i & 1];
        v16h bfr[4];
#pragma unroll
        for (int nt = 0; nt < 4; ++nt)
            bfr[nt] = load_b_f16(bs + (nt * 16 + l15) * 32 + 8 * lhalf);
#pragma unroll
        for (int nt = 0; nt < 4; ++nt)
            acc[nt] = wmma16(a, bfr[nt], acc[nt]);
        __syncthreads();
    }
#pragma unroll
    for (int nt = 0; nt < 4; ++nt) {
        const int n = nbase + nt * 16 + l15;
        const float bvv = bias[n];
#pragma unroll
        for (int r = 0; r < 8; ++r) {
            const int m = mbase + r + 8 * lhalf;
            out[(size_t)m * Dd + n] = acc[nt][r] + bvv;
        }
    }
}

extern "C" void kernel_launch(void* const* d_in, const int* in_sizes, int n_in,
                              void* d_out, int out_size, void* d_ws, size_t ws_size,
                              hipStream_t stream) {
    const float* query = (const float*)d_in[0];
    const float* key   = (const float*)d_in[1];
    const float* value = (const float*)d_in[2];
    const float* Wq    = (const float*)d_in[3];
    const float* bq    = (const float*)d_in[4];
    const float* Wk    = (const float*)d_in[5];
    const float* bk    = (const float*)d_in[6];
    const float* Wv    = (const float*)d_in[7];
    const float* bv    = (const float*)d_in[8];
    const float* Wo    = (const float*)d_in[9];
    const float* bo    = (const float*)d_in[10];

    float* out  = (float*)d_out;
    float* attn = out + (size_t)BS * Dd;           // [B,H,S,S]

    const size_t nQKV = (size_t)Bb * Hh * Ss * DHh; // 4,194,304 f16 elements each
    _Float16* Qh  = (_Float16*)d_ws;
    _Float16* Kh  = Qh + nQKV;
    _Float16* Vth = Kh + nQKV;                      // V transposed [B,H,DH,S]
    _Float16* Oh  = Vth + nQKV;                     // context [B,S,D] f16
    float* stMax  = (float*)(Oh + nQKV);            // [B*H*S]
    float* stInv  = stMax + (size_t)Bb * Hh * Ss;

    dim3 blk(256);
    dim3 gp(Dd / 64, BS / 128);    // (16, 32) — projection GEMMs
    dim3 ga(Ss / 128, Bb * Hh);    // (16, 32) — attention phases

    proj_kernel<<<gp, blk, 0, stream>>>(query, Wq, bq, Qh, 0);
    proj_kernel<<<gp, blk, 0, stream>>>(key,   Wk, bk, Kh, 0);
    proj_kernel<<<gp, blk, 0, stream>>>(value, Wv, bv, Vth, 1);
    scores_kernel<<<ga, blk, 0, stream>>>(Qh, Kh, attn, stMax, stInv);
    attnv_kernel<<<ga, blk, 0, stream>>>(attn, Vth, stMax, stInv, Oh);
    outproj_kernel<<<gp, blk, 0, stream>>>(Oh, Wo, bo, out);
}
